// LearnableGaussianCrossAttention_15453292331456
// MI455X (gfx1250) — matrix operand order repaired and checked by reading, not compiled, not run on hardware
//
#include <hip/hip_runtime.h>

// ---------------------------------------------------------------------------
// LearnableGaussianCrossAttention for MI455X (gfx1250, wave32, WMMA)
// B=2, N_M=1024, N_O=4096, C=512, H=8, D=64
// Bandwidth-bound on the 256MB attn output (write + read + write ~ 0.77 GB
// @ 23.3 TB/s); all GEMMs in f16 WMMA with f32 accumulate.
// ---------------------------------------------------------------------------

typedef __attribute__((ext_vector_type(16))) _Float16 v16h;
typedef __attribute__((ext_vector_type(8)))  float    v8f;
typedef __attribute__((ext_vector_type(2)))  _Float16 h2;

constexpr int BB = 2;
constexpr int NM = 1024;
constexpr int NO = 4096;
constexpr int CC = 512;
constexpr int HH = 8;
constexpr int DD = 64;

// ---------------------------------------------------------------------------
// fp32 -> fp16 elementwise convert (for weight matrices)
// ---------------------------------------------------------------------------
__global__ void f32_to_f16_kernel(const float* __restrict__ src,
                                  _Float16* __restrict__ dst, int n) {
    int i = blockIdx.x * blockDim.x + threadIdx.x;
    if (i < n) dst[i] = (_Float16)src[i];
}

// ---------------------------------------------------------------------------
// mean of channel_scores over last dim -> avgP[b]
// ---------------------------------------------------------------------------
__global__ void mean_kernel(const float* __restrict__ cs,
                            float* __restrict__ avg, int n) {
    int b = blockIdx.x;
    float s = 0.f;
    for (int i = threadIdx.x; i < n; i += blockDim.x) s += cs[(size_t)b * n + i];
    __shared__ float sm[256];
    sm[threadIdx.x] = s;
    __syncthreads();
    for (int k = 128; k > 0; k >>= 1) {
        if (threadIdx.x < k) sm[threadIdx.x] += sm[threadIdx.x + k];
        __syncthreads();
    }
    if (threadIdx.x == 0) avg[b] = sm[0] / (float)n;
}

// ---------------------------------------------------------------------------
// Generic GEMM: out[M,N] = A[M,K] @ W[N,K]^T (+bias), WMMA f32<-f16.
// Block = 256 threads = 8 waves; each wave owns a 16x16 tile; block tile 32x64.
// A-fragment lane layout (16-bit A 16x32): m = lane&15, half = lane>>4,
//   VGPR v packs K = (v>=4?16:0) + half*8 + (v&3)*2 , +1
// B-fragment (32x16): n = lane&15, K same as A-fragment
// D-fragment: row = v + 8*half, col = lane&15
// OUT_TRANS stores f16 transposed: out[col*M + row] -> each lane writes 8
// contiguous halfs (one 16B store), and downstream B-fragment loads of this
// tensor become contiguous b128s per lane.
// ---------------------------------------------------------------------------
template <bool A_IS_HALF, bool OUT_HALF, bool OUT_TRANS>
__global__ void gemm_xWt_kernel(const void* __restrict__ Ap,
                                const _Float16* __restrict__ W,
                                int M, int N, int K,
                                void* __restrict__ outp,
                                const float* __restrict__ bias) {
    const int lane = threadIdx.x & 31;
    const int w    = threadIdx.x >> 5;
    const int mtiles  = M >> 5;
    const int bm      = blockIdx.x % mtiles;
    const int bn      = blockIdx.x / mtiles;
    const int rowbase = bm * 32 + (w & 1) * 16;
    const int colbase = bn * 64 + (w >> 1) * 16;
    const int m  = lane & 15;
    const int kh = lane >> 4;

    v8f acc = {};
    for (int k0 = 0; k0 < K; k0 += 32) {
        v16h a, b;
#pragma unroll
        for (int v = 0; v < 8; ++v) {
            const int ka = ((v & 4) << 2) + kh * 8 + (v & 3) * 2;
            if constexpr (A_IS_HALF) {
                const _Float16* A = (const _Float16*)Ap;
                h2 av = *(const h2*)&A[(size_t)(rowbase + m) * K + k0 + ka];
                a[2 * v] = av.x; a[2 * v + 1] = av.y;
            } else {
                const float* A = (const float*)Ap;
                float2 av = *(const float2*)&A[(size_t)(rowbase + m) * K + k0 + ka];
                a[2 * v] = (_Float16)av.x; a[2 * v + 1] = (_Float16)av.y;
            }
            const int kb = ((v & 4) << 2) + kh * 8 + (v & 3) * 2;
            h2 bv = *(const h2*)&W[(size_t)(colbase + m) * K + k0 + kb];
            b[2 * v] = bv.x; b[2 * v + 1] = bv.y;
        }
        acc = __builtin_amdgcn_wmma_f32_16x16x32_f16(false, a, false, b,
                                                     (short)0, acc, false, false);
    }
#pragma unroll
    for (int v = 0; v < 8; ++v) {
        const int row = rowbase + v + 8 * kh;
        const int col = colbase + m;
        if constexpr (OUT_TRANS) {
            ((_Float16*)outp)[(size_t)col * M + row] = (_Float16)acc[v];
        } else if constexpr (OUT_HALF) {
            ((_Float16*)outp)[(size_t)row * N + col] = (_Float16)acc[v];
        } else {
            float bv = bias ? bias[col] : 0.0f;
            ((float*)outp)[(size_t)row * N + col] = acc[v] + bv;
        }
    }
}

// ---------------------------------------------------------------------------
// Fused fuzzy attention. One block per (b, h, 16-row tile of N_M).
// Pass 1: scores = Q_h K_h^T * scale + avgP ; e = exp(exp(-(s-mu)^2/(2 sig^2)))
//         store e into attn output; accumulate per-row sums in LDS.
// Pass 2: normalize attn in place (required output) while consuming it as the
//         A matrix of attn @ V_h; V is consumed from the transposed layout
//         Vt[C][B*N_O] so B-fragments are contiguous 16B loads per lane.
// Prefetches are unconditional (speculative prefetch: overshoot is silently
// dropped / lands in adjacent workspace) so no EXEC masking in the hot loop.
// All control flow is wave-uniform (EXEC == all ones around every WMMA).
// ---------------------------------------------------------------------------
__global__ void fuzzy_attn_kernel(const _Float16* __restrict__ Qh,
                                  const _Float16* __restrict__ Kh,
                                  const _Float16* __restrict__ Vt,
                                  const float* __restrict__ avgP,
                                  const float* __restrict__ mu,
                                  const float* __restrict__ sigma,
                                  float* __restrict__ attn,
                                  _Float16* __restrict__ Oh) {
    const int lane = threadIdx.x & 31;
    const int w    = threadIdx.x >> 5;
    const int rt   = blockIdx.x & (NM / 16 - 1);          // 64 row tiles
    const int h    = (blockIdx.x >> 6) & (HH - 1);
    const int b    = blockIdx.x >> 9;
    const int rowbase = rt * 16;
    const int m  = lane & 15;
    const int kh = lane >> 4;

    __shared__ float s_red[16];
    __shared__ float s_out[16 * 64];

    if (threadIdx.x < 16) s_red[threadIdx.x] = 0.f;
#pragma unroll
    for (int i = 0; i < 4; ++i) s_out[threadIdx.x + i * 256] = 0.f;
    __syncthreads();

    // --- load Q A-fragments once (16 x 64 -> two k-steps of 32) ---
    v16h aq[2];
#pragma unroll
    for (int s = 0; s < 2; ++s) {
#pragma unroll
        for (int v = 0; v < 8; ++v) {
            const int ka = s * 32 + ((v & 4) << 2) + kh * 8 + (v & 3) * 2;
            h2 qv = *(const h2*)&Qh[((size_t)b * NM + rowbase + m) * CC + h * DD + ka];
            aq[s][2 * v] = qv.x; aq[s][2 * v + 1] = qv.y;
        }
    }

    const float scale  = 0.125f;                 // D^-0.5
    const float aP     = avgP[b];
    const float muh    = mu[h];
    const float sg     = sigma[h] + 1e-8f;
    const float inv2s2 = 1.0f / (2.0f * sg * sg);

    float* attn_bh = attn + (((size_t)b * HH + h) * NM + rowbase) * NO;

    float psum[8];
#pragma unroll
    for (int v = 0; v < 8; ++v) psum[v] = 0.f;

    // -------- Pass 1: scores -> e = exp(fuzzy); row sums --------
    for (int ct = w; ct < NO / 16; ct += 8) {
        const int colbase = ct * 16;
        // speculative prefetch of this wave's next K column-tile
        __builtin_prefetch(
            &Kh[((size_t)b * NO + colbase + 128 + m) * CC + h * DD], 0, 1);
        v8f acc = {};
#pragma unroll
        for (int s = 0; s < 2; ++s) {
            v16h bk;
#pragma unroll
            for (int v = 0; v < 8; ++v) {
                const int kb = s * 32 + ((v & 4) << 2) + kh * 8 + (v & 3) * 2;
                h2 kv2 = *(const h2*)&Kh[((size_t)b * NO + colbase + m) * CC + h * DD + kb];
                bk[2 * v] = kv2.x; bk[2 * v + 1] = kv2.y;
            }
            acc = __builtin_amdgcn_wmma_f32_16x16x32_f16(false, aq[s], false, bk,
                                                         (short)0, acc, false, false);
        }
#pragma unroll
        for (int v = 0; v < 8; ++v) {
            const int row = v + 8 * kh;
            float sc = acc[v] * scale + aP;
            float d  = sc - muh;
            float fz = __expf(-d * d * inv2s2);
            float e  = __expf(fz);
            attn_bh[(size_t)row * NO + colbase + m] = e;
            psum[v] += e;
        }
    }
#pragma unroll
    for (int v = 0; v < 8; ++v) atomicAdd(&s_red[v + 8 * kh], psum[v]);
    __syncthreads();
    if (threadIdx.x < 16) s_red[threadIdx.x] = 1.0f / s_red[threadIdx.x];
    __syncthreads();

    // -------- Pass 2: normalize attn in place; O = attn @ V_h --------
    const float rinv = s_red[m];                 // A-fragment row = m
    v8f oacc[4] = {{}, {}, {}, {}};
    for (int kt = w; kt < NO / 32; kt += 8) {
        const int kbase = kt * 32;
        // speculative prefetch of this wave's next Vt stripe
        __builtin_prefetch(
            &Vt[((size_t)(h * DD + m)) * (BB * NO) + (size_t)b * NO + kbase + 256], 0, 1);
        v16h a;
#pragma unroll
        for (int v = 0; v < 8; ++v) {
            const int ka  = ((v & 4) << 2) + kh * 8 + (v & 3) * 2;
            const size_t idx = (size_t)m * NO + kbase + ka;
            float e0 = attn_bh[idx]     * rinv;
            float e1 = attn_bh[idx + 1] * rinv;
            attn_bh[idx]     = e0;               // final normalized attn output
            attn_bh[idx + 1] = e1;
            a[2 * v] = (_Float16)e0; a[2 * v + 1] = (_Float16)e1;
        }
#pragma unroll
        for (int g = 0; g < 4; ++g) {
            v16h bv;
#pragma unroll
            for (int v = 0; v < 8; ++v) {
                const int kb = ((v & 4) << 2) + kh * 8 + (v & 3) * 2;
                h2 vv = *(const h2*)&Vt[((size_t)(h * DD + g * 16 + m)) * (BB * NO) +
                                        (size_t)b * NO + kbase + kb];
                bv[2 * v] = vv.x; bv[2 * v + 1] = vv.y;
            }
            oacc[g] = __builtin_amdgcn_wmma_f32_16x16x32_f16(false, a, false, bv,
                                                             (short)0, oacc[g],
                                                             false, false);
        }
    }
#pragma unroll
    for (int g = 0; g < 4; ++g)
#pragma unroll
        for (int v = 0; v < 8; ++v)
            atomicAdd(&s_out[(v + 8 * kh) * 64 + g * 16 + m], oacc[g][v]);
    __syncthreads();

#pragma unroll
    for (int i = 0; i < 4; ++i) {
        const int idx = threadIdx.x + i * 256;
        const int row = idx >> 6, col = idx & 63;
        Oh[((size_t)b * NM + rowbase + row) * CC + h * DD + col] =
            (_Float16)s_out[idx];
    }
}

// ---------------------------------------------------------------------------
// Host-side launch
// ---------------------------------------------------------------------------
extern "C" void kernel_launch(void* const* d_in, const int* in_sizes, int n_in,
                              void* d_out, int out_size, void* d_ws, size_t ws_size,
                              hipStream_t stream) {
    (void)in_sizes; (void)n_in; (void)out_size; (void)ws_size;

    const float* q_merged = (const float*)d_in[0];
    const float* kv_orig  = (const float*)d_in[1];
    const float* chs      = (const float*)d_in[2];
    const float* Wq       = (const float*)d_in[3];
    const float* Wk       = (const float*)d_in[4];
    const float* Wv       = (const float*)d_in[5];
    const float* Wo       = (const float*)d_in[6];
    const float* bo       = (const float*)d_in[7];
    const float* mu       = (const float*)d_in[8];
    const float* sigma    = (const float*)d_in[9];

    char*  ws  = (char*)d_ws;
    size_t off = 0;
    auto alloc = [&](size_t bytes) -> void* {
        void* p = ws + off;
        off = (off + bytes + 255) & ~(size_t)255;
        return p;
    };

    _Float16* Wq_h = (_Float16*)alloc((size_t)CC * CC * 2);
    _Float16* Wk_h = (_Float16*)alloc((size_t)CC * CC * 2);
    _Float16* Wv_h = (_Float16*)alloc((size_t)CC * CC * 2);
    _Float16* Wo_h = (_Float16*)alloc((size_t)CC * CC * 2);
    _Float16* Qh   = (_Float16*)alloc((size_t)BB * NM * CC * 2);
    _Float16* Kh   = (_Float16*)alloc((size_t)BB * NO * CC * 2);
    _Float16* Vt   = (_Float16*)alloc((size_t)BB * NO * CC * 2);  // [C][B*NO]
    _Float16* Oh   = (_Float16*)alloc((size_t)BB * NM * CC * 2);
    float*    avgP = (float*)alloc(BB * sizeof(float));

    float* out  = (float*)d_out;
    float* attn = out + (size_t)BB * NM * CC;   // second tuple output

    // weight conversion fp32 -> fp16
    const int nW = CC * CC;
    f32_to_f16_kernel<<<(nW + 255) / 256, 256, 0, stream>>>(Wq, Wq_h, nW);
    f32_to_f16_kernel<<<(nW + 255) / 256, 256, 0, stream>>>(Wk, Wk_h, nW);
    f32_to_f16_kernel<<<(nW + 255) / 256, 256, 0, stream>>>(Wv, Wv_h, nW);
    f32_to_f16_kernel<<<(nW + 255) / 256, 256, 0, stream>>>(Wo, Wo_h, nW);

    // avg_P per batch
    mean_kernel<<<BB, 256, 0, stream>>>(chs, avgP, NO);

    // projections (f16 outputs; V transposed to [C][B*NO])
    const int Mq = BB * NM, Mkv = BB * NO;
    gemm_xWt_kernel<false, true, false>
        <<<(Mq / 32) * (CC / 64), 256, 0, stream>>>(q_merged, Wq_h, Mq, CC, CC, Qh, nullptr);
    gemm_xWt_kernel<false, true, false>
        <<<(Mkv / 32) * (CC / 64), 256, 0, stream>>>(kv_orig, Wk_h, Mkv, CC, CC, Kh, nullptr);
    gemm_xWt_kernel<false, true, true>
        <<<(Mkv / 32) * (CC / 64), 256, 0, stream>>>(kv_orig, Wv_h, Mkv, CC, CC, Vt, nullptr);

    // fused fuzzy attention (writes normalized attn to d_out + O to ws)
    fuzzy_attn_kernel<<<BB * HH * (NM / 16), 256, 0, stream>>>(
        Qh, Kh, Vt, avgP, mu, sigma, attn, Oh);

    // output projection (f32 + bias) into d_out
    gemm_xWt_kernel<true, false, false>
        <<<(Mq / 32) * (CC / 64), 256, 0, stream>>>(Oh, Wo_h, Mq, CC, CC, out, bo);
}
